// LGF_16922171146835
// MI455X (gfx1250) — matrix operand order repaired and checked by reading, not compiled, not run on hardware
//
#include <hip/hip_runtime.h>

// ---------------------------------------------------------------------------
// Deep Guided Filter, fused for gfx1250 (wave32, WMMA, async-LDS, TDM).
//   K1: guided_map = conv1x1(relu(conv1x1(img,w1,b1)),w2,b2)  via v_wmma f16
//       (w2 uploaded to LDS via Tensor Data Mover when available)
//   K2: per-(b,ch) 32x32-tile fused guided filter; x-tile staged into LDS
//       via global_load_async_to_lds_b128 when available.
// ---------------------------------------------------------------------------

typedef __attribute__((ext_vector_type(16))) _Float16 v16h;
typedef __attribute__((ext_vector_type(8)))  float    v8f;
typedef int v4i_ __attribute__((vector_size(16)));   // matches builtin proto

#define AS3 __attribute__((address_space(3)))

#ifndef __has_builtin
#define __has_builtin(x) 0
#endif

#if __has_builtin(__builtin_amdgcn_global_load_async_to_lds_b128)
#define HAS_ASYNC_LDS 1
#else
#define HAS_ASYNC_LDS 0
#endif

#if __has_builtin(__builtin_amdgcn_s_wait_asynccnt)
#define WAIT_ASYNC0() __builtin_amdgcn_s_wait_asynccnt(0)
#else
#define WAIT_ASYNC0() asm volatile("s_wait_asynccnt 0" ::: "memory")
#endif

#if __has_builtin(__builtin_amdgcn_tensor_load_to_lds) && \
    __has_builtin(__builtin_amdgcn_s_wait_tensorcnt)
#define HAS_TDM 1
#if __has_include(<hip/amd_detail/amd_gfx1250_TDM.h>)
#define TDM_6ARG 1
#else
#define TDM_6ARG 0
#endif
#else
#define HAS_TDM 0
#endif

#define B_    2
#define H_    512
#define W_    1024
#define HID   64
#define COUT  19
#define LH    64
#define LW    128

// ============================ Kernel 1: convs ==============================
// Each wave handles 64 consecutive pixels, 16 per step (M rows of WMMA). A = hidden
// activations (16 pixels x 64, f16), B = w2^T (64 x 19 padded to 2 N-tiles).
// Wave32 fragment layouts per CDNA5 ISA 7.12.2.
__global__ __launch_bounds__(256) void dgf_conv_kernel(
    const float* __restrict__ img,
    const float* __restrict__ w1, const float* __restrict__ b1,
    const float* __restrict__ w2, const float* __restrict__ b2,
    float* __restrict__ xout)   // [B][COUT][H][W]
{
  __shared__ float sw1[HID * 3];
  __shared__ float sb1[HID];
  __shared__ __align__(16) float sw2[COUT * HID];
  __shared__ float sb2[COUT];

  const int tid = threadIdx.x;
  for (int i = tid; i < HID * 3; i += 256) sw1[i] = w1[i];
  for (int i = tid; i < HID;     i += 256) sb1[i] = b1[i];
  for (int i = tid; i < COUT;    i += 256) sb2[i] = b2[i];

#if HAS_TDM
  // TDM upload of w2 (2D f32 tensor 64 x 19, full tile) into LDS.
  // D# per CDNA5 ISA.
  typedef unsigned int u32x4 __attribute__((ext_vector_type(4)));
  typedef int          i32x4 __attribute__((ext_vector_type(4)));
  typedef int          i32x8 __attribute__((ext_vector_type(8)));
  if (tid < 32) {
    const unsigned long long ga = (unsigned long long)(__UINTPTR_TYPE__)w2;
    const unsigned lds_off = (unsigned)(__UINTPTR_TYPE__)(AS3 void*)&sw2[0];
    u32x4 g0;
    g0[0] = 1u;                                            // count=1, user mode
    g0[1] = lds_off;                                       // lds_addr (bytes)
    g0[2] = (unsigned)(ga & 0xffffffffull);                // global_addr lo
    g0[3] = (unsigned)((ga >> 32) & 0x1ffffffull) | (2u << 30);  // hi | type=2
    i32x8 g1;
    const unsigned td0 = HID, td1 = COUT;                  // tensor dims (elts)
    const unsigned s0 = HID, s1 = HID * COUT;              // strides (elts)
    g1[0] = (int)(2u << 16);                               // data_size=4B
    g1[1] = (int)((td0 & 0xffffu) << 16);                  // tensor_dim0 lo16
    g1[2] = (int)((td0 >> 16) | ((td1 & 0xffffu) << 16));  // dim0 hi | dim1 lo
    g1[3] = (int)((td1 >> 16) | (HID << 16));              // dim1 hi | tile_dim0
    g1[4] = (int)(COUT);                                   // tile_dim1 | tile_dim2=0
    g1[5] = (int)s0;                                       // dim0_stride lo32
    g1[6] = (int)((s1 & 0xffffu) << 16);                   // s0 hi16=0 | s1 lo16
    g1[7] = (int)(s1 >> 16);                               // s1 hi32
    i32x4 gz = {0, 0, 0, 0};
#if TDM_6ARG
    i32x8 gz8 = {0, 0, 0, 0, 0, 0, 0, 0};
    __builtin_amdgcn_tensor_load_to_lds(g0, g1, gz, gz, gz8, 0);
#else
    __builtin_amdgcn_tensor_load_to_lds(g0, g1, gz, gz, 0);
#endif
    __builtin_amdgcn_s_wait_tensorcnt(0);
  }
#else
  for (int i = tid; i < COUT * HID; i += 256) sw2[i] = w2[i];
#endif
  __syncthreads();

  const int wave = tid >> 5;
  const int lane = tid & 31;
  const int half = lane >> 4;
  const int l    = lane & 15;

  // B fragments for w2: bf[n_tile][k_step]; lane l holds column (l), K=half*8+i (i<8) else 8+half*8+i.
  v16h bf[2][2];
#pragma unroll
  for (int nt = 0; nt < 2; ++nt) {
    const int n = nt * 16 + l;
#pragma unroll
    for (int ks = 0; ks < 2; ++ks) {
#pragma unroll
      for (int i = 0; i < 16; ++i) {
        const int K = ks * 32 + ((i < 8) ? (half * 8 + i) : (8 + half * 8 + i));
        const float v = (n < COUT) ? sw2[n * HID + K] : 0.0f;
        bf[nt][ks][i] = (_Float16)v;
      }
    }
  }

#pragma unroll 1
  for (int g4 = 0; g4 < 4; ++g4) {
    const long p0  = (long)blockIdx.x * 512 + wave * 64 + g4 * 16;
    const int  bb  = (int)(p0 / (long)(H_ * W_));
    const int  rem = (int)(p0 % (long)(H_ * W_));
    const int  hrow = rem / W_;
    const int  w0   = rem % W_;

    const long ibase = ((long)(bb * 3) * H_ + hrow) * W_ + w0 + l;
    const float r  = img[ibase];
    const float g  = img[ibase + (long)H_ * W_];
    const float bl = img[ibase + 2L * H_ * W_];
    if (g4 < 3) __builtin_prefetch(&img[ibase + 16], 0, 0);  // global_prefetch_b8

    // Hidden layer (K=3 conv + relu) directly into A-fragment layout:
    // lane l row l; a0 elem i -> K = half*8+i (i<8) else 8+half*8+i; a1 -> +32.
    v16h a0 = {}, a1 = {};
#pragma unroll
    for (int i = 0; i < 16; ++i) {
      const int K0 = (i < 8) ? (half * 8 + i) : (8 + half * 8 + i);
      {
        const int o = K0;
        float hv = sb1[o] + r * sw1[o * 3 + 0] + g * sw1[o * 3 + 1] + bl * sw1[o * 3 + 2];
        a0[i] = (_Float16)(hv > 0.0f ? hv : 0.0f);
      }
      {
        const int o = 32 + K0;
        float hv = sb1[o] + r * sw1[o * 3 + 0] + g * sw1[o * 3 + 1] + bl * sw1[o * 3 + 2];
        a1[i] = (_Float16)(hv > 0.0f ? hv : 0.0f);
      }
    }

    v8f c0 = {}, c1 = {};
    c0 = __builtin_amdgcn_wmma_f32_16x16x32_f16(false, a0, false, bf[0][0], (short)0, c0, false, false);
    c0 = __builtin_amdgcn_wmma_f32_16x16x32_f16(false, a1, false, bf[0][1], (short)0, c0, false, false);
    c1 = __builtin_amdgcn_wmma_f32_16x16x32_f16(false, a0, false, bf[1][0], (short)0, c1, false, false);
    c1 = __builtin_amdgcn_wmma_f32_16x16x32_f16(false, a1, false, bf[1][1], (short)0, c1, false, false);

    // Store D: lane holds channel (l [+16]) at pixel columns v + 8*half.
    {
      const int ch = l;
      const float bias = sb2[ch];
      const long base = ((long)(bb * COUT + ch) * H_ + hrow) * W_ + w0 + half * 8;
#pragma unroll
      for (int v = 0; v < 8; ++v) xout[base + v] = c0[v] + bias;
    }
    {
      const int ch = 16 + l;
      if (ch < COUT) {
        const float bias = sb2[ch];
        const long base = ((long)(bb * COUT + ch) * H_ + hrow) * W_ + w0 + half * 8;
#pragma unroll
        for (int v = 0; v < 8; ++v) xout[base + v] = c1[v] + bias;
      }
    }
  }
}

// ========================= Kernel 2: guided filter =========================
#define RR     4
#define TILE   32
#define FRAME  48            // TILE + 4*RR (double halo)
#define FXP    48            // sx stride (16B-aligned rows for async b128)
#define FYP    49            // sy stride (padded)
#define MID    40            // TILE + 2*RR
#define MP     41
#define HAP    33

__global__ __launch_bounds__(256) void dgf_gf_kernel(
    const float* __restrict__ x,      // guided map [B][COUT][H][W]
    const float* __restrict__ ylow,   // [B][COUT][LH][LW]
    float* __restrict__ out)          // [B][COUT][H][W]
{
  // Float pool with phase-overlapped views (peak 56,672 B):
  //  [0     .. 2304)  sx  [48][48]      (live whole kernel)
  //  [2304  .. 4656)  sy  [48][49]      -> later A [40][41]
  //  [4656  .. 12528) Hx/Hy/Hxy/Hxx [48][41] -> later HA/HB [40][33]
  //  [12528 .. 14168) Bb  [40][41]
  __shared__ __align__(16) float pool[14168];
  float* const sx  = pool;
  float* const sy  = pool + 2304;
  float* const Hx  = pool + 4656;
  float* const Hy  = Hx + 1968;
  float* const Hxy = Hy + 1968;
  float* const Hxx = Hxy + 1968;
  float* const A_  = pool + 2304;    // overlays sy (dead after phase 2)
  float* const Bb  = pool + 12528;
  float* const HA  = pool + 4656;    // overlays Hx.. (dead after phase 3)
  float* const HB  = HA + 1320;

  const int tid = threadIdx.x;
  const int tx0 = blockIdx.x * TILE;
  const int ty0 = blockIdx.y * TILE;
  const int bc  = blockIdx.z;                 // b*COUT + ch
  const float* __restrict__ xsrc = x    + (long)bc * H_ * W_;
  const float* __restrict__ ysrc = ylow + (long)bc * LH * LW;

  const float scy = (float)(LH - 1) / (float)(H_ - 1);
  const float scx = (float)(LW - 1) / (float)(W_ - 1);

  // ---- Phase 1a: x tile(+8 halo) into LDS ----
#if HAS_ASYNC_LDS
  // 4-float chunks are always fully in- or out-of-image (borders % 4 == 0).
  for (int idx = tid; idx < FRAME * (FRAME / 4); idx += 256) {
    const int i = idx / (FRAME / 4);
    const int j = (idx % (FRAME / 4)) * 4;
    const int gy = ty0 - 8 + i, gx = tx0 - 8 + j;
    if (gy >= 0 && gy < H_ && gx >= 0 && gx + 3 < W_) {
      __builtin_amdgcn_global_load_async_to_lds_b128(
          (v4i_*)(xsrc + (long)gy * W_ + gx),
          (v4i_*)&sx[i * FXP + j], 0, 0);
    } else {
      sx[i * FXP + j + 0] = 0.0f;
      sx[i * FXP + j + 1] = 0.0f;
      sx[i * FXP + j + 2] = 0.0f;
      sx[i * FXP + j + 3] = 0.0f;
    }
  }
#else
  for (int idx = tid; idx < FRAME * FRAME; idx += 256) {
    const int i = idx / FRAME, j = idx % FRAME;
    const int gy = ty0 - 8 + i, gx = tx0 - 8 + j;
    float xv = 0.0f;
    if (gy >= 0 && gy < H_ && gx >= 0 && gx < W_) xv = xsrc[(long)gy * W_ + gx];
    sx[i * FXP + j] = xv;
  }
#endif

  // ---- Phase 1b: bilinear-upsample y (align_corners) into LDS ----
  for (int idx = tid; idx < FRAME * FRAME; idx += 256) {
    const int i = idx / FRAME, j = idx % FRAME;
    const int gy = ty0 - 8 + i, gx = tx0 - 8 + j;
    float yv = 0.0f;
    if (gy >= 0 && gy < H_ && gx >= 0 && gx < W_) {
      const float fy = gy * scy, fx = gx * scx;
      int y0 = (int)fy, x0 = (int)fx;
      const int y1 = (y0 + 1 < LH) ? y0 + 1 : LH - 1;
      const int x1 = (x0 + 1 < LW) ? x0 + 1 : LW - 1;
      const float wy = fy - (float)y0, wx = fx - (float)x0;
      const float t00 = ysrc[y0 * LW + x0], t01 = ysrc[y0 * LW + x1];
      const float t10 = ysrc[y1 * LW + x0], t11 = ysrc[y1 * LW + x1];
      yv = (t00 * (1.0f - wx) + t01 * wx) * (1.0f - wy)
         + (t10 * (1.0f - wx) + t11 * wx) * wy;
    }
    sy[i * FYP + j] = yv;
  }
#if HAS_ASYNC_LDS
  WAIT_ASYNC0();
#endif
  __syncthreads();

  // ---- Phase 2: horizontal 9-sums of x, y, xy, xx at frame cols 4..43 ----
  for (int idx = tid; idx < FRAME * MID; idx += 256) {
    const int i = idx / MID, j = idx % MID;
    float ax = 0.f, ay = 0.f, axy = 0.f, axx = 0.f;
#pragma unroll
    for (int k = 0; k <= 2 * RR; ++k) {
      const float xv = sx[i * FXP + j + k];
      const float yv = sy[i * FYP + j + k];
      ax += xv; ay += yv; axy += xv * yv; axx += xv * xv;
    }
    Hx[i * MP + j]  = ax;  Hy[i * MP + j]  = ay;
    Hxy[i * MP + j] = axy; Hxx[i * MP + j] = axx;
  }
  __syncthreads();

  // ---- Phase 3: vertical 9-sums -> means -> A, b at tile+4 halo ----
  // (zero off-image; truncated-window counts N computed analytically)
  for (int idx = tid; idx < MID * MID; idx += 256) {
    const int ii = idx / MID, jj = idx % MID;
    const int gy = ty0 - 4 + ii, gx = tx0 - 4 + jj;
    float Av = 0.0f, Bv = 0.0f;
    if (gy >= 0 && gy < H_ && gx >= 0 && gx < W_) {
      float ax = 0.f, ay = 0.f, axy = 0.f, axx = 0.f;
#pragma unroll
      for (int k = 0; k <= 2 * RR; ++k) {
        ax  += Hx[(ii + k) * MP + jj];
        ay  += Hy[(ii + k) * MP + jj];
        axy += Hxy[(ii + k) * MP + jj];
        axx += Hxx[(ii + k) * MP + jj];
      }
      const int ny = ((gy + RR < H_) ? gy + RR : H_ - 1) - ((gy - RR > 0) ? gy - RR : 0) + 1;
      const int nx = ((gx + RR < W_) ? gx + RR : W_ - 1) - ((gx - RR > 0) ? gx - RR : 0) + 1;
      const float invn = 1.0f / (float)(ny * nx);
      const float mx = ax * invn, my = ay * invn;
      const float cov = axy * invn - mx * my;
      const float var = axx * invn - mx * mx;
      Av = cov / (var + 0.01f);
      Bv = my - Av * mx;
    }
    A_[ii * MP + jj] = Av;
    Bb[ii * MP + jj] = Bv;
  }
  __syncthreads();

  // ---- Phase 4: horizontal 9-sums of A, b ----
  for (int idx = tid; idx < MID * TILE; idx += 256) {
    const int i = idx / TILE, j = idx % TILE;
    float sa = 0.f, sb = 0.f;
#pragma unroll
    for (int k = 0; k <= 2 * RR; ++k) {
      sa += A_[i * MP + j + k];
      sb += Bb[i * MP + j + k];
    }
    HA[i * HAP + j] = sa;
    HB[i * HAP + j] = sb;
  }
  __syncthreads();

  // ---- Phase 5: vertical 9-sums -> mean_A, mean_b -> output ----
  for (int idx = tid; idx < TILE * TILE; idx += 256) {
    const int i = idx / TILE, j = idx % TILE;
    float sa = 0.f, sb = 0.f;
#pragma unroll
    for (int k = 0; k <= 2 * RR; ++k) {
      sa += HA[(i + k) * HAP + j];
      sb += HB[(i + k) * HAP + j];
    }
    const int gy = ty0 + i, gx = tx0 + j;
    const int ny = ((gy + RR < H_) ? gy + RR : H_ - 1) - ((gy - RR > 0) ? gy - RR : 0) + 1;
    const int nx = ((gx + RR < W_) ? gx + RR : W_ - 1) - ((gx - RR > 0) ? gx - RR : 0) + 1;
    const float invn = 1.0f / (float)(ny * nx);
    const float xv = sx[(i + 8) * FXP + (j + 8)];
    out[((long)bc * H_ + gy) * W_ + gx] = (sa * invn) * xv + (sb * invn);
  }
}

// ================================ Launch ===================================
extern "C" void kernel_launch(void* const* d_in, const int* in_sizes, int n_in,
                              void* d_out, int out_size, void* d_ws, size_t ws_size,
                              hipStream_t stream) {
  const float* img    = (const float*)d_in[0];  // [2,3,512,1024]
  const float* outlow = (const float*)d_in[1];  // [2,19,64,128]
  const float* w1     = (const float*)d_in[2];  // [64,3]
  const float* b1     = (const float*)d_in[3];  // [64]
  const float* w2     = (const float*)d_in[4];  // [19,64]
  const float* b2     = (const float*)d_in[5];  // [19]
  float* out  = (float*)d_out;                  // [2,19,512,1024]
  float* xbuf = (float*)d_ws;                   // guided map scratch (~76 MB)

  // K1: 2*512*1024 pixels / (512 px per 256-thread block) = 2048 blocks.
  dgf_conv_kernel<<<2048, 256, 0, stream>>>(img, w1, b1, w2, b2, xbuf);

  // K2: 32x16 spatial tiles x (B*COUT)=38 planes.
  dgf_gf_kernel<<<dim3(W_ / TILE, H_ / TILE, B_ * COUT), 256, 0, stream>>>(
      xbuf, outlow, out);
}
